// DotProductAttentionWithRPR_16183436772080
// MI455X (gfx1250) — matrix operand order repaired and checked by reading, not compiled, not run on hardware
//
#include <hip/hip_runtime.h>

// ---------------------------------------------------------------------------
// RPR attention for MI455X (gfx1250), wave32, WMMA bf16.
//   B=64, L=1024, D=64
// 3-kernel pipeline; bf16 intermediates in workspace:
//   ws[0        .. 128MB) : Spos  bf16  [q][b][k]   (pos scores)
//   ws[128MB    .. 256MB) : attnW bf16  [b][q][k]   (softmax probs)
//   ws[256MB    .. 272MB) : Ocont f32   [b][q][d]   (content output)
// ---------------------------------------------------------------------------

#define Lk 1024
#define Bn 64
#define Dn 64

typedef __attribute__((ext_vector_type(16))) __bf16          v16bf;
typedef __attribute__((ext_vector_type(16))) unsigned short  v16u;
typedef __attribute__((ext_vector_type(8)))  float           v8f;

// Native bf16 converts -> backend emits v_cvt_*bf16 instead of bit-twiddling.
__device__ __forceinline__ unsigned short f2bf(float f) {
  union { __bf16 b; unsigned short u; } v;
  v.b = (__bf16)f;
  return v.u;
}
__device__ __forceinline__ float bf2f(unsigned short h) {
  union { unsigned short u; __bf16 b; } v;
  v.u = h;
  return (float)v.b;
}

__device__ __forceinline__ v8f wmma_bf16(v16u a, v16u b, v8f c) {
  union { v16u u; v16bf b; } A, B; A.u = a; B.u = b;
  return __builtin_amdgcn_wmma_f32_16x16x32_bf16(
      false, A.b, false, B.b, (short)0, c, false, false);
}

// A operand (16x32 bf16), source LDS row-major [m][k], contraction contiguous.
// ISA layout: lane m=l&15; half=l>>4; VGPR r pair at k = 16*(r>>2)+8*half+2*(r&3).
__device__ __forceinline__ v16u load_A_bf(const unsigned short* mat, int ld,
                                          int m0, int k0) {
  const int lane = threadIdx.x & 31;
  const unsigned short* row = mat + (m0 + (lane & 15)) * ld + k0;
  const int half = (lane >> 4) << 3;
  v16u a;
#pragma unroll
  for (int r = 0; r < 8; ++r) {
    int k = ((r >> 2) << 4) + half + ((r & 3) << 1);
    unsigned pr = *(const unsigned*)(row + k);
    a[2 * r]     = (unsigned short)pr;
    a[2 * r + 1] = (unsigned short)(pr >> 16);
  }
  return a;
}

// Same as load_A_bf but source is f32 LDS (converted on the fly).
__device__ __forceinline__ v16u load_A_f32(const float* mat, int ld,
                                           int m0, int k0) {
  const int lane = threadIdx.x & 31;
  const float* row = mat + (m0 + (lane & 15)) * ld + k0;
  const int half = (lane >> 4) << 3;
  v16u a;
#pragma unroll
  for (int r = 0; r < 8; ++r) {
    int k = ((r >> 2) << 4) + half + ((r & 3) << 1);
    a[2 * r]     = f2bf(row[k]);
    a[2 * r + 1] = f2bf(row[k + 1]);
  }
  return a;
}

// B operand (32x16 bf16). Source LDS laid out [n][k] (contraction contiguous).
// ISA layout: lane n=l&15; half=l>>4; elem j<8 at K 8*half+j, j>=8 at K 16+8*half+(j-8).
__device__ __forceinline__ v16u load_B_bf(const unsigned short* matT, int ld,
                                          int n0, int k0) {
  const int lane = threadIdx.x & 31;
  const unsigned short* row =
      matT + (n0 + (lane & 15)) * ld + k0 + ((lane >> 4) << 3);
  v16u b;
#pragma unroll
  for (int j = 0; j < 16; ++j) b[j] = row[j < 8 ? j : j + 8];
  return b;
}

// ---------------------------------------------------------------------------
// Kernel 1: Spos[q][b][k] = sum_d Q[b][q][d] * posK[q][k][d]
// grid (8, 1024): one WG = one q, one 128-wide k panel. GEMM 64x128x64.
// ---------------------------------------------------------------------------
__global__ void k1_pos_scores(const float* __restrict__ Q,
                              const float* __restrict__ posK,
                              unsigned short* __restrict__ Spos) {
  const int q = blockIdx.y, k0 = blockIdx.x * 128;
  const int t = threadIdx.x, wave = t >> 5, lane = t & 31;
  __shared__ __align__(16) unsigned short Qs[64 * 64];   // [b][d] bf16
  __shared__ __align__(16) unsigned short Ks[128 * 64];  // [k][d] bf16

#pragma unroll
  for (int i = 0; i < 16; ++i) {                 // 4096 elems
    int e = i * 256 + t;
    int bb = e >> 6, d = e & 63;
    Qs[e] = f2bf(Q[((size_t)bb * Lk + q) * Dn + d]);
  }
#pragma unroll
  for (int i = 0; i < 32; ++i) {                 // 8192 elems
    int e = i * 256 + t;
    int kk = e >> 6, d = e & 63;
    Ks[e] = f2bf(posK[((size_t)q * Lk + k0 + kk) * Dn + d]);
  }
  // streaming prefetch of the next pos_k panel (global_prefetch_b8)
  if (k0 + 128 < Lk)
    __builtin_prefetch(posK + ((size_t)q * Lk + k0 + 128) * Dn + (t << 4), 0, 1);
  __syncthreads();

  const int m0 = (wave >> 1) * 16;               // b-tile
#pragma unroll
  for (int j = 0; j < 4; ++j) {
    const int n0 = ((wave & 1) * 4 + j) * 16;    // k-tile within panel
    v8f c = {};
#pragma unroll
    for (int ks = 0; ks < 64; ks += 32) {
      v16u a  = load_A_bf(Qs, 64, m0, ks);
      v16u bm = load_B_bf(Ks, 64, n0, ks);
      c = wmma_bf16(a, bm, c);
    }
#pragma unroll
    for (int r = 0; r < 8; ++r) {
      int m = m0 + r + ((lane >> 4) << 3);       // batch index
      int n = k0 + n0 + (lane & 15);             // key index
      Spos[((size_t)q * Bn + m) * Lk + n] = f2bf(c[r]);
    }
  }
}

// ---------------------------------------------------------------------------
// Kernel 2: per (b, 16-query tile): content scores + pos + mask + softmax +
//           out_content = attn*V.   grid (64, 64), block 256 (8 waves).
// Dynamic LDS: S f32[16*1024] (64KB) + KV bf16[128*64] (16KB) = 80KB.
// ---------------------------------------------------------------------------
__global__ void k2_attn(const float* __restrict__ Q, const float* __restrict__ K,
                        const float* __restrict__ V, const int* __restrict__ vlens,
                        const unsigned short* __restrict__ Spos,
                        unsigned short* __restrict__ attnW,
                        float* __restrict__ Ocont) {
  const int b = blockIdx.y, q0 = blockIdx.x * 16;
  const int t = threadIdx.x, wave = t >> 5, lane = t & 31;
  extern __shared__ char smem[];
  float* S = (float*)smem;                                      // [16][1024] f32
  unsigned short* KV = (unsigned short*)(smem + 16 * Lk * 4);   // 128x64 bf16
  __shared__ __align__(16) unsigned short Qs[16 * 64];
  __shared__ __align__(16) float Osh[16 * 64];
  __shared__ float red[16 * 16];
  __shared__ float rowmax[16], rowsum[16];

  const int vlen = vlens[b];
  const float scale = 0.125f;                    // 1/sqrt(64)

#pragma unroll
  for (int i = 0; i < 4; ++i) {                  // Q tile -> bf16
    int e = i * 256 + t;
    int m = e >> 6, d = e & 63;
    Qs[e] = f2bf(Q[((size_t)b * Lk + q0 + m) * Dn + d]);
  }
#pragma unroll
  for (int i = 0; i < 4; ++i) Osh[i * 256 + t] = 0.f;

  // ---- scores: S = (Qtile*K^T + Spos)*scale, masked -----------------------
  for (int p = 0; p < 8; ++p) {
    const int kp0 = p * 128;
    __syncthreads();
#pragma unroll
    for (int i = 0; i < 32; ++i) {               // K panel [kk][d]
      int e = i * 256 + t;
      int kk = e >> 6, d = e & 63;
      KV[e] = f2bf(K[((size_t)b * Lk + kp0 + kk) * Dn + d]);
    }
    __syncthreads();
    const int n0 = wave * 16;
    v8f c = {};
#pragma unroll
    for (int ks = 0; ks < 64; ks += 32) {
      v16u a  = load_A_bf(Qs, 64, 0, ks);
      v16u bm = load_B_bf(KV, 64, n0, ks);
      c = wmma_bf16(a, bm, c);
    }
    const int ncol = kp0 + n0 + (lane & 15);
#pragma unroll
    for (int r = 0; r < 8; ++r) {
      int m = r + ((lane >> 4) << 3);
      float sp = bf2f(Spos[((size_t)(q0 + m) * Bn + b) * Lk + ncol]);
      float v = (c[r] + sp) * scale;
      if (ncol >= vlen) v = -1.0e6f;
      S[m * Lk + ncol] = v;
    }
  }
  __syncthreads();

  // ---- softmax over k (rows resident in LDS) ------------------------------
  {
    const int r = t >> 4, s = t & 15;
    float mx = -3.4e38f;
    for (int k = s; k < Lk; k += 16) mx = fmaxf(mx, S[r * Lk + k]);
    red[r * 16 + s] = mx;
    __syncthreads();
    if (s == 0) {
      float m2 = red[r * 16];
#pragma unroll
      for (int j = 1; j < 16; ++j) m2 = fmaxf(m2, red[r * 16 + j]);
      rowmax[r] = m2;
    }
    __syncthreads();
    const float rm = rowmax[r];
    float sum = 0.f;
    for (int k = s; k < Lk; k += 16) {
      float e = __expf(S[r * Lk + k] - rm);
      S[r * Lk + k] = e;
      sum += e;
    }
    red[r * 16 + s] = sum;
    __syncthreads();
    if (s == 0) {
      float s2 = 0.f;
#pragma unroll
      for (int j = 0; j < 16; ++j) s2 += red[r * 16 + j];
      rowsum[r] = s2;
    }
    __syncthreads();
  }
  // normalize in LDS + spill attn (bf16) for kernel 3, coalesced
  for (int i = 0; i < 64; ++i) {
    int e = i * 256 + t;                         // e = m*1024 + k
    int m = e >> 10, k = e & 1023;
    float a = S[e] * (1.f / rowsum[m]);
    S[e] = a;
    attnW[((size_t)b * Lk + q0 + m) * Lk + k] = f2bf(a);
  }

  // ---- out_content = attn(16x1024) * V(1024x64) ---------------------------
  {
    const int nt = wave & 3, khalf = wave >> 2;  // waves split N and K
    const int n0 = nt * 16;
    v8f c = {};
    for (int p = 0; p < 8; ++p) {
      const int kp0 = p * 128;
      __syncthreads();
#pragma unroll
      for (int i = 0; i < 32; ++i) {             // V panel transposed: Vt[d][kk]
        int e = i * 256 + t;
        int kk = e >> 6, d = e & 63;
        KV[d * 128 + kk] = f2bf(V[((size_t)b * Lk + kp0 + kk) * Dn + d]);
      }
      __syncthreads();
#pragma unroll
      for (int ks = 0; ks < 64; ks += 32) {
        int kw = khalf * 64 + ks;
        v16u a  = load_A_f32(S, Lk, 0, kp0 + kw);
        v16u bm = load_B_bf(KV, 128, n0, kw);
        c = wmma_bf16(a, bm, c);
      }
    }
#pragma unroll
    for (int r = 0; r < 8; ++r) {                // ds_add_f32 cross-wave reduce
      int m = r + ((lane >> 4) << 3);
      atomicAdd(&Osh[m * 64 + n0 + (lane & 15)], c[r]);
    }
  }
  __syncthreads();
#pragma unroll
  for (int i = 0; i < 4; ++i) {
    int e = i * 256 + t;
    int m = e >> 6, d = e & 63;
    Ocont[((size_t)b * Lk + q0 + m) * Dn + d] = Osh[e];
  }
}

// ---------------------------------------------------------------------------
// Kernel 3: per q: out[b][q][:] = attn[b][q][:]*posV[q] + Ocont[b][q][:]
// GEMM M=64(b) x N=64(d) x K=1024.  grid 1024, block 256.
// attn panel staged with CDNA5 async copy: global_load_async_to_lds_b128.
// ---------------------------------------------------------------------------
__global__ void k3_pos_out(const unsigned short* __restrict__ attnW,
                           const float* __restrict__ posV,
                           const float* __restrict__ Ocont,
                           float* __restrict__ out) {
  const int q = blockIdx.x;
  const int t = threadIdx.x, wave = t >> 5, lane = t & 31;
  __shared__ __align__(16) unsigned short As[64 * 128];  // attn [b][k]
  __shared__ __align__(16) unsigned short Vt[64 * 128];  // posV^T [d][k]
  const int mt = wave >> 1;
  const int ntb = (wave & 1) * 2;
  v8f c0 = {}, c1 = {};

  for (int p = 0; p < 8; ++p) {
    const int kp0 = p * 128;
    __syncthreads();
    // attn panel: pure bf16 copy, 16KB -> async DMA to LDS (ASYNCcnt path).
    // 1024 chunks of 16B; per-lane LDS addr + per-lane global offset (GVS mode).
    {
      const unsigned short* gbase = attnW + (size_t)q * Lk + kp0;  // uniform base
#pragma unroll
      for (int i = 0; i < 4; ++i) {
        int c = i * 256 + t;
        int bb = c >> 4, j = c & 15;
        unsigned voff = (unsigned)(bb * (Lk * Lk * 2) + j * 16);   // bytes
        unsigned lds  = (unsigned)(size_t)&As[bb * 128 + j * 8];   // LDS byte addr
        asm volatile("global_load_async_to_lds_b128 %0, %1, %2 offset:0"
                     :: "v"(lds), "v"(voff), "s"(gbase)
                     : "memory");
      }
    }
#pragma unroll
    for (int i = 0; i < 32; ++i) {               // posV panel, transposed to [d][k]
      int e = i * 256 + t;
      int kk = e >> 6, d = e & 63;
      Vt[d * 128 + kk] = f2bf(posV[((size_t)q * Lk + kp0 + kk) * Dn + d]);
    }
    if (kp0 + 128 < Lk)
      __builtin_prefetch(posV + ((size_t)q * Lk + kp0 + 128) * Dn + (t << 4), 0, 1);
    asm volatile("s_wait_asynccnt 0x0" ::: "memory");  // my async writes visible
    __syncthreads();                                    // everyone's visible
#pragma unroll
    for (int ks = 0; ks < 128; ks += 32) {
      v16u a = load_A_bf(As, 128, mt * 16, ks);
      c0 = wmma_bf16(a, load_B_bf(Vt, 128, (ntb + 0) * 16, ks), c0);
      c1 = wmma_bf16(a, load_B_bf(Vt, 128, (ntb + 1) * 16, ks), c1);
    }
  }
#pragma unroll
  for (int r = 0; r < 8; ++r) {
    int m = mt * 16 + r + ((lane >> 4) << 3);    // batch index
    int n = ntb * 16 + (lane & 15);              // d index
    size_t o0 = ((size_t)m * Lk + q) * Dn;
    out[o0 + n]      = c0[r] + Ocont[o0 + n];
    out[o0 + n + 16] = c1[r] + Ocont[o0 + n + 16];
  }
}

// ---------------------------------------------------------------------------
extern "C" void kernel_launch(void* const* d_in, const int* in_sizes, int n_in,
                              void* d_out, int out_size, void* d_ws, size_t ws_size,
                              hipStream_t stream) {
  (void)in_sizes; (void)n_in; (void)out_size; (void)ws_size;
  const float* Q    = (const float*)d_in[0];
  const float* K    = (const float*)d_in[1];
  const float* V    = (const float*)d_in[2];
  const float* posK = (const float*)d_in[3];
  const float* posV = (const float*)d_in[4];
  const int*   vl   = (const int*)d_in[5];
  float* out = (float*)d_out;

  char* ws = (char*)d_ws;
  unsigned short* Spos  = (unsigned short*)ws;                        // 128 MB
  unsigned short* attnW = (unsigned short*)(ws + (size_t)134217728);  // 128 MB
  float*          Ocont = (float*)(ws + (size_t)268435456);           //  16 MB

  k1_pos_scores<<<dim3(8, 1024), 256, 0, stream>>>(Q, posK, Spos);

  size_t sm2 = (size_t)16 * Lk * sizeof(float) + (size_t)128 * 64 * sizeof(unsigned short);
  k2_attn<<<dim3(64, 64), 256, sm2, stream>>>(Q, K, V, vl, Spos, attnW, Ocont);

  k3_pos_out<<<dim3(1024), 256, 0, stream>>>(attnW, posV, Ocont, out);
}
